// LongformerBiLSTMForTokenClassification_87875030876975
// MI455X (gfx1250) — matrix operand order repaired and checked by reading, not compiled, not run on hardware
//
#include <hip/hip_runtime.h>
#include <hip/hip_bf16.h>

// ---------------- model constants ----------------
#define Bb   2
#define Ss   4096
#define Dd   768
#define Hh   12
#define DH   64
#define Ww   256
#define Cc   (Ss / Ww)     // 16 chunks
#define FFd  3072
#define Ll   4
#define EMBd (4 * Dd)      // 3072
#define HIDd 200
#define GATE (4 * HIDd)    // 800

typedef __attribute__((ext_vector_type(16))) __bf16 v16bf;
typedef __attribute__((ext_vector_type(8)))  __bf16 v8bf;
typedef __attribute__((ext_vector_type(8)))  float  v8f;

// float -> bf16 (hardware cvt, RNE)
__device__ __forceinline__ __bf16 f2bf(float f) { return (__bf16)f; }

// pack two floats into two bf16 in one dword: exactly one v_cvt_pk_bf16_f32.
// Inline asm keeps the SLP vectorizer from pairing lanes across calls and
// generating half-shuffle cleanup code.
__device__ __forceinline__ unsigned pack2bf(float a, float b) {
    unsigned r;
    asm("v_cvt_pk_bf16_f32 %0, %1, %2" : "=v"(r) : "v"(a), "v"(b));
    return r;
}

__device__ __forceinline__ v8f zero8() {
    v8f z;
#pragma unroll
    for (int e = 0; e < 8; ++e) z[e] = 0.f;
    return z;
}

__device__ __forceinline__ v16bf cat8(v8bf lo, v8bf hi) {
    return __builtin_shufflevector(lo, hi, 0, 1, 2, 3, 4, 5, 6, 7,
                                           8, 9, 10, 11, 12, 13, 14, 15);
}
// 16 contiguous bf16 (two ds_load_b128)
__device__ __forceinline__ v16bf ldsB(const __bf16* p) {
    v8bf lo = *(const v8bf*)p;
    v8bf hi = *(const v8bf*)(p + 8);
    return cat8(lo, hi);
}
// A-operand: elements 0..15 at row[16*half .. 16*half+15] (contiguous)
__device__ __forceinline__ v16bf ldsA(const __bf16* row, int half) {
    v8bf lo = *(const v8bf*)(row + 16 * half);
    v8bf hi = *(const v8bf*)(row + 16 * half + 8);
    return cat8(lo, hi);
}

__device__ __forceinline__ v8f wmma_bf16(v16bf a, v16bf b, v8f c) {
    // D = A(16x32 bf16) * B(32x16 bf16) + C(16x16 f32)
    return __builtin_amdgcn_wmma_f32_16x16x32_bf16(false, a, false, b, (short)0, c, false, false);
}

// reductions across the 16 lanes of a half-wave (xor 1,2,4,8 keeps lane>>4)
__device__ __forceinline__ float rmax16(float v) {
#pragma unroll
    for (int o = 1; o < 16; o <<= 1) v = fmaxf(v, __shfl_xor(v, o, 32));
    return v;
}
__device__ __forceinline__ float rsum16(float v) {
#pragma unroll
    for (int o = 1; o < 16; o <<= 1) v += __shfl_xor(v, o, 32);
    return v;
}

// ---------------- embedding gather ----------------
__global__ __launch_bounds__(256) void embed_kernel(const int* __restrict__ ids,
                                                    const float* __restrict__ we,
                                                    const float* __restrict__ pe,
                                                    float* __restrict__ X) {
    int r = blockIdx.x;            // b*S + s
    int s = r % Ss;
    int id = ids[r];
    for (int i = threadIdx.x; i < Dd; i += 256)
        X[(size_t)r * Dd + i] = we[(size_t)id * Dd + i] + pe[(size_t)s * Dd + i];
}

// ---------------- residual add + LayerNorm (optionally scatter into EMB concat) ----
__global__ __launch_bounds__(256) void addln_kernel(const float* __restrict__ xin,
                                                    const float* __restrict__ y,
                                                    const float* __restrict__ g,
                                                    const float* __restrict__ bta,
                                                    float* __restrict__ xout,
                                                    float* __restrict__ embout, int embOff) {
    __shared__ float row[Dd];
    __shared__ float red[256];
    int r = blockIdx.x, tid = threadIdx.x;
    float s = 0.f;
    for (int i = tid; i < Dd; i += 256) {
        float v = xin[(size_t)r * Dd + i];
        if (y) v += y[(size_t)r * Dd + i];
        row[i] = v; s += v;
    }
    red[tid] = s; __syncthreads();
    for (int o = 128; o > 0; o >>= 1) { if (tid < o) red[tid] += red[tid + o]; __syncthreads(); }
    float mean = red[0] * (1.f / Dd);
    __syncthreads();
    float vs = 0.f;
    for (int i = tid; i < Dd; i += 256) { float d = row[i] - mean; vs += d * d; }
    red[tid] = vs; __syncthreads();
    for (int o = 128; o > 0; o >>= 1) { if (tid < o) red[tid] += red[tid + o]; __syncthreads(); }
    float rstd = rsqrtf(red[0] * (1.f / Dd) + 1e-5f);
    for (int i = tid; i < Dd; i += 256) {
        float v = (row[i] - mean) * rstd * g[i] + bta[i];
        xout[(size_t)r * Dd + i] = v;
        if (embout) embout[(size_t)r * EMBd + embOff + i] = v;
    }
}

// ---------------- generic bf16 WMMA GEMM: C[M,N] = act(A[M,K] @ B + bias) -------
// Block tile 128(M) x 64(N); 8 waves, each computing a 32x32 tile (4 WMMA / k-step).
// REQUIRES: M % 128 == 0, K % 32 == 0 (true for all call sites; N may be ragged).
// transB==0: B is [K,N] row-major ; transB==1: B is [N,K] row-major (used as B^T)
// act: 0 none, 1 exact GELU
__global__ __launch_bounds__(256) void gemm_kernel(const float* __restrict__ A,
                                                   const float* __restrict__ Bm,
                                                   const float* __restrict__ bias,
                                                   float* __restrict__ Cm,
                                                   int M, int N, int Kd, int transB, int act) {
    __shared__ alignas(16) __bf16 As[128][32];    // [m][k]
    __shared__ alignas(16) __bf16 BsT[64][32];    // [n][k]  (transposed stage)
    const int tid = threadIdx.x;
    const int lane = tid & 31, wave = tid >> 5;
    const int half = lane >> 4, l15 = lane & 15;
    const int mi = wave >> 1, ni = wave & 1;      // 4x(M32) by 2x(N32)
    const int m0 = blockIdx.y * 128, n0 = blockIdx.x * 64;

    v8f acc00 = zero8(), acc01 = zero8(), acc10 = zero8(), acc11 = zero8();

    for (int kb = 0; kb < Kd; kb += 32) {
        __syncthreads();
        // ---- stage A tile 128x32 : 4 float4 per thread, b64 LDS stores (no M guard) ----
#pragma unroll
        for (int q = 0; q < 4; ++q) {
            int idx4 = tid + 256 * q;             // 0..1023
            int r = idx4 >> 3, c4 = (idx4 & 7) * 4;
            float4 v = *(const float4*)(A + (size_t)(m0 + r) * Kd + kb + c4);
            *(uint2*)&As[r][c4] = make_uint2(pack2bf(v.x, v.y), pack2bf(v.z, v.w));
        }
        // ---- stage B tile 32(k) x 64(n) into BsT[n][k] ----
        if (transB) {
#pragma unroll
            for (int q = 0; q < 2; ++q) {
                int idx4 = tid + 256 * q;         // 0..511
                int cc = idx4 >> 3, k4 = (idx4 & 7) * 4;
                int gn = n0 + cc;
                float4 v = make_float4(0.f, 0.f, 0.f, 0.f);
                if (gn < N) v = *(const float4*)(Bm + (size_t)gn * Kd + kb + k4);
                *(uint2*)&BsT[cc][k4] = make_uint2(pack2bf(v.x, v.y), pack2bf(v.z, v.w));
            }
        } else {
#pragma unroll
            for (int q = 0; q < 2; ++q) {
                int idx = tid + 256 * q;          // 0..511
                int cc = idx & 63, k4 = (idx >> 6) * 4;
                int gn = n0 + cc;
                float v0 = 0.f, v1 = 0.f, v2 = 0.f, v3 = 0.f;
                if (gn < N) {
                    const float* bp = Bm + (size_t)(kb + k4) * N + gn;
                    v0 = bp[0]; v1 = bp[(size_t)N]; v2 = bp[2 * (size_t)N]; v3 = bp[3 * (size_t)N];
                }
                *(uint2*)&BsT[cc][k4] = make_uint2(pack2bf(v0, v1), pack2bf(v2, v3));
            }
        }
        // prefetch next K tiles into cache while this one is consumed
        int nkb = kb + 32;
        if (nkb < Kd) {
            int r = tid >> 1, ccp = (tid & 1) * 16;
            __builtin_prefetch(A + (size_t)(m0 + r) * Kd + nkb + ccp, 0, 3);
            int gn = n0 + (tid & 63);
            if (gn < N && tid < 64) {
                const float* pb = transB ? (Bm + (size_t)gn * Kd + nkb)
                                         : (Bm + (size_t)nkb * N + gn);
                __builtin_prefetch(pb, 0, 3);
            }
        }
        __syncthreads();

        v16bf a0 = ldsA(&As[32 * mi + l15][0], half);
        v16bf a1 = ldsA(&As[32 * mi + 16 + l15][0], half);
        v16bf b0 = ldsB(&BsT[32 * ni + l15][16 * half]);
        v16bf b1 = ldsB(&BsT[32 * ni + 16 + l15][16 * half]);
        acc00 = wmma_bf16(a0, b0, acc00);
        acc01 = wmma_bf16(a0, b1, acc01);
        acc10 = wmma_bf16(a1, b0, acc10);
        acc11 = wmma_bf16(a1, b1, acc11);
    }

#pragma unroll
    for (int ti = 0; ti < 2; ++ti) {
#pragma unroll
        for (int tj = 0; tj < 2; ++tj) {
            v8f acc = ti ? (tj ? acc11 : acc10) : (tj ? acc01 : acc00);
            int gn = n0 + 32 * ni + 16 * tj + l15;
            if (gn < N) {
                float bv = bias ? bias[gn] : 0.f;
#pragma unroll
                for (int e = 0; e < 8; ++e) {
                    int gm = m0 + 32 * mi + 16 * ti + e + 8 * half;
                    float v = acc[e] + bv;
                    if (act == 1) v = 0.5f * v * (1.f + erff(v * 0.70710678118654752f));
                    Cm[(size_t)gm * N + gn] = v;
                }
            }
        }
    }
}

// ---------------- sliding-window attention, flash style, one block per (b,h,chunk)
__global__ __launch_bounds__(256) void attn_kernel(const float* __restrict__ Q,
                                                   const float* __restrict__ K,
                                                   const float* __restrict__ V,
                                                   const int* __restrict__ amask,
                                                   float* __restrict__ O) {
    __shared__ alignas(16) __bf16 ks[64][64];     // [key][dh]   (B-operand for q.k^T)
    __shared__ alignas(16) __bf16 vst[64][64];    // [dh][key]   (B-operand for p.v)
    __shared__ alignas(16) __bf16 ps[8][16][64];  // per-wave P tile (16 q rows x 64 keys)

    const int blk = blockIdx.x;
    const int c = blk & (Cc - 1);
    const int h = (blk >> 4) % Hh;
    const int b = blk / (Cc * Hh);
    const int tid = threadIdx.x;
    const int lane = tid & 31, wave = tid >> 5;
    const int half = lane >> 4, l15 = lane & 15;

    // q tile in A-operand layout, pre-scaled by 1/sqrt(DH); vector loads + packed cvt
    v16bf qa0, qa1;
    {
        const float* qr = Q + ((size_t)(b * Ss + c * Ww + wave * 16 + l15)) * Dd + h * DH;
        union { v16bf v; unsigned u[8]; } a0, a1;
#pragma unroll
        for (int g = 0; g < 2; ++g) {             // g=0 -> elements 0..7, g=1 -> 8..15
            const float* p0 = qr + 16 * half + 8 * g;        // qa0 chunk
            const float* p1 = qr + 32 + 16 * half + 8 * g;   // qa1 chunk
            float4 x0 = *(const float4*)p0;
            float4 x1 = *(const float4*)(p0 + 4);
            float4 y0 = *(const float4*)p1;
            float4 y1 = *(const float4*)(p1 + 4);
            a0.u[4 * g + 0] = pack2bf(x0.x * 0.125f, x0.y * 0.125f);
            a0.u[4 * g + 1] = pack2bf(x0.z * 0.125f, x0.w * 0.125f);
            a0.u[4 * g + 2] = pack2bf(x1.x * 0.125f, x1.y * 0.125f);
            a0.u[4 * g + 3] = pack2bf(x1.z * 0.125f, x1.w * 0.125f);
            a1.u[4 * g + 0] = pack2bf(y0.x * 0.125f, y0.y * 0.125f);
            a1.u[4 * g + 1] = pack2bf(y0.z * 0.125f, y0.w * 0.125f);
            a1.u[4 * g + 2] = pack2bf(y1.x * 0.125f, y1.y * 0.125f);
            a1.u[4 * g + 3] = pack2bf(y1.z * 0.125f, y1.w * 0.125f);
        }
        qa0 = a0.v; qa1 = a1.v;
    }

    float oldm[8], lrun[8];
    v8f oc[4];
#pragma unroll
    for (int e = 0; e < 8; ++e) { oldm[e] = -1e30f; lrun[e] = 0.f; }
#pragma unroll
    for (int od = 0; od < 4; ++od) oc[od] = zero8();

    for (int t = 0; t < 12; ++t) {       // 12 key tiles of 64 cover the 3w=768 window
        __syncthreads();
#pragma unroll
        for (int q = 0; q < 4; ++q) {    // cooperative K/V tile load (float4, zero-fill OOB)
            int idx4 = tid + 256 * q;    // 0..1023
            int r = idx4 >> 4;           // key row 0..63
            int c4 = (idx4 & 15) * 4;    // dh 0,4,...,60
            int kpos = (c - 1) * Ww + t * 64 + r;
            float4 kv = make_float4(0.f, 0.f, 0.f, 0.f);
            float4 vv = make_float4(0.f, 0.f, 0.f, 0.f);
            if (kpos >= 0 && kpos < Ss) {
                size_t base = ((size_t)(b * Ss + kpos)) * Dd + h * DH + c4;
                kv = *(const float4*)(K + base);
                vv = *(const float4*)(V + base);
            }
            *(uint2*)&ks[r][c4] = make_uint2(pack2bf(kv.x, kv.y), pack2bf(kv.z, kv.w));
            vst[c4 + 0][r] = f2bf(vv.x);
            vst[c4 + 1][r] = f2bf(vv.y);
            vst[c4 + 2][r] = f2bf(vv.z);
            vst[c4 + 3][r] = f2bf(vv.w);
        }
        __syncthreads();

        // scores S = q * k^T  (16 rows x 64 keys per wave)
        v8f sj[4];
#pragma unroll
        for (int j = 0; j < 4; ++j) {
            v8f s = zero8();
            const __bf16* kr = &ks[16 * j + l15][0];
            s = wmma_bf16(qa0, ldsB(kr + 16 * half), s);
            s = wmma_bf16(qa1, ldsB(kr + 32 + 16 * half), s);
            sj[j] = s;
        }

        // mask (band + key-valid) and row maxima
        unsigned okb[4];
        float tmax[8];
#pragma unroll
        for (int e = 0; e < 8; ++e) tmax[e] = -1e30f;
#pragma unroll
        for (int j = 0; j < 4; ++j) {
            okb[j] = 0u;
            int jw = t * 64 + 16 * j + l15;
            int kpos = (c - 1) * Ww + jw;
            bool inb = (kpos >= 0) && (kpos < Ss);
            bool kvld = inb && (amask[b * Ss + (inb ? kpos : 0)] != 0);
#pragma unroll
            for (int e = 0; e < 8; ++e) {
                int qi = wave * 16 + e + 8 * half;
                bool ok = kvld && (jw >= qi) && (jw <= qi + 2 * Ww);
                float val = ok ? sj[j][e] : -1e9f;
                if (ok) okb[j] |= (1u << e);
                sj[j][e] = val;
                tmax[e] = fmaxf(tmax[e], val);
            }
        }

        float newm[8], alpha[8], lsum[8];
#pragma unroll
        for (int e = 0; e < 8; ++e) {
            float tm = rmax16(tmax[e]);
            float nm = fmaxf(oldm[e], tm);
            newm[e] = nm;
            alpha[e] = __expf(oldm[e] - nm);
            lsum[e] = 0.f;
        }
#pragma unroll
        for (int j = 0; j < 4; ++j) {
#pragma unroll
            for (int e = 0; e < 8; ++e) {
                bool ok = (okb[j] >> e) & 1u;
                float p = ok ? __expf(sj[j][e] - newm[e]) : 0.f;
                ps[wave][e + 8 * half][16 * j + l15] = f2bf(p);
                lsum[e] += p;
            }
        }
#pragma unroll
        for (int e = 0; e < 8; ++e) {
            lrun[e] = lrun[e] * alpha[e] + rsum16(lsum[e]);
            oldm[e] = newm[e];
        }
#pragma unroll
        for (int od = 0; od < 4; ++od)
#pragma unroll
            for (int e = 0; e < 8; ++e) oc[od][e] *= alpha[e];

        __syncthreads();   // make wave-written P visible for cross-lane A-layout reload

        // O += P * V   (A = P re-read in A layout; B = transposed V, contiguous)
        const __bf16* pr = &ps[wave][l15][0];
        v16bf am0 = ldsA(pr, half);
        v16bf am1 = ldsA(pr + 32, half);
#pragma unroll
        for (int od = 0; od < 4; ++od) {
            const __bf16* vr = &vst[16 * od + l15][0];
            oc[od] = wmma_bf16(am0, ldsB(vr + 16 * half), oc[od]);
            oc[od] = wmma_bf16(am1, ldsB(vr + 32 + 16 * half), oc[od]);
        }
    }

#pragma unroll
    for (int od = 0; od < 4; ++od)
#pragma unroll
        for (int e = 0; e < 8; ++e) {
            int qi = wave * 16 + e + 8 * half;
            O[((size_t)(b * Ss + c * Ww + qi)) * Dd + h * DH + 16 * od + l15] = oc[od][e] / lrun[e];
        }
}

// ---------------- serial LSTM recurrence (one block per direction) ----------------
__global__ __launch_bounds__(256) void lstm_kernel(const float* __restrict__ xz,   // [B,S,800], bias included
                                                   const float* __restrict__ Whh,  // [800,200]
                                                   float* __restrict__ Hout,       // [B,S,200]
                                                   int reverse) {
    __shared__ alignas(16) float hsh[2 * HIDd];
    __shared__ float csh[2 * HIDd];
    __shared__ float hnew[2 * HIDd];
    int tid = threadIdx.x;
    for (int i = tid; i < 2 * HIDd; i += 256) { hsh[i] = 0.f; csh[i] = 0.f; }
    __syncthreads();
    for (int step = 0; step < Ss; ++step) {
        int t = reverse ? (Ss - 1 - step) : step;
        for (int idx = tid; idx < 2 * HIDd; idx += 256) {
            int bb = idx / HIDd, j = idx % HIDd;
            const float* hv = &hsh[bb * HIDd];
            const float* xzr = xz + ((size_t)(bb * Ss + t)) * GATE;
            float zi = xzr[j], zf = xzr[HIDd + j], zg = xzr[2 * HIDd + j], zo = xzr[3 * HIDd + j];
            const float* wi = Whh + (size_t)j * HIDd;
            const float* wf = Whh + (size_t)(HIDd + j) * HIDd;
            const float* wg = Whh + (size_t)(2 * HIDd + j) * HIDd;
            const float* wo = Whh + (size_t)(3 * HIDd + j) * HIDd;
#pragma unroll 2
            for (int k = 0; k < HIDd; k += 4) {
                float4 hk = *(const float4*)(hv + k);
                float4 a = *(const float4*)(wi + k);
                float4 bq = *(const float4*)(wf + k);
                float4 cq = *(const float4*)(wg + k);
                float4 dq = *(const float4*)(wo + k);
                zi += hk.x * a.x + hk.y * a.y + hk.z * a.z + hk.w * a.w;
                zf += hk.x * bq.x + hk.y * bq.y + hk.z * bq.z + hk.w * bq.w;
                zg += hk.x * cq.x + hk.y * cq.y + hk.z * cq.z + hk.w * cq.w;
                zo += hk.x * dq.x + hk.y * dq.y + hk.z * dq.z + hk.w * dq.w;
            }
            float ig = 1.f / (1.f + __expf(-zi));
            float fg = 1.f / (1.f + __expf(-zf));
            float og = 1.f / (1.f + __expf(-zo));
            float cv = fg * csh[idx] + ig * tanhf(zg);
            float hn = og * tanhf(cv);
            csh[idx] = cv;
            hnew[idx] = hn;
            Hout[((size_t)(bb * Ss + t)) * HIDd + j] = hn;
        }
        __syncthreads();
        for (int i = tid; i < 2 * HIDd; i += 256) hsh[i] = hnew[i];
        __syncthreads();
    }
}

// ---------------- classifier ----------------
__global__ __launch_bounds__(256) void cls_kernel(const float* __restrict__ HF,
                                                  const float* __restrict__ HB,
                                                  const float* __restrict__ Wc,  // [400,2]
                                                  const float* __restrict__ bc,
                                                  float* __restrict__ out) {
    int r = blockIdx.x * blockDim.x + threadIdx.x;
    if (r >= Bb * Ss) return;
    float a0 = bc[0], a1 = bc[1];
    const float* hf = HF + (size_t)r * HIDd;
    const float* hb = HB + (size_t)r * HIDd;
    for (int k = 0; k < HIDd; ++k) {
        a0 += hf[k] * Wc[k * 2 + 0] + hb[k] * Wc[(HIDd + k) * 2 + 0];
        a1 += hf[k] * Wc[k * 2 + 1] + hb[k] * Wc[(HIDd + k) * 2 + 1];
    }
    out[(size_t)r * 2 + 0] = a0;
    out[(size_t)r * 2 + 1] = a1;
}

// ---------------- launcher ----------------
extern "C" void kernel_launch(void* const* d_in, const int* in_sizes, int n_in,
                              void* d_out, int out_size, void* d_ws, size_t ws_size,
                              hipStream_t stream) {
    const int*   ids      = (const int*)d_in[0];
    const int*   amask    = (const int*)d_in[1];
    const float* word_emb = (const float*)d_in[2];
    const float* pos_emb  = (const float*)d_in[3];
    const float* emb_g    = (const float*)d_in[4];
    const float* emb_b    = (const float*)d_in[5];
    const float* Wq = (const float*)d_in[6];   const float* bq = (const float*)d_in[7];
    const float* Wk = (const float*)d_in[8];   const float* bk = (const float*)d_in[9];
    const float* Wv = (const float*)d_in[10];  const float* bv = (const float*)d_in[11];
    const float* Wo = (const float*)d_in[12];  const float* bo = (const float*)d_in[13];
    const float* g1 = (const float*)d_in[14];  const float* b1 = (const float*)d_in[15];
    const float* Wf1 = (const float*)d_in[16]; const float* bf1 = (const float*)d_in[17];
    const float* Wf2 = (const float*)d_in[18]; const float* bf2 = (const float*)d_in[19];
    const float* g2 = (const float*)d_in[20];  const float* b2 = (const float*)d_in[21];
    const float* Wih_f = (const float*)d_in[22]; const float* Whh_f = (const float*)d_in[23];
    const float* b_f   = (const float*)d_in[24];
    const float* Wih_b = (const float*)d_in[25]; const float* Whh_b = (const float*)d_in[26];
    const float* b_b   = (const float*)d_in[27];
    const float* clsW  = (const float*)d_in[28]; const float* clsb = (const float*)d_in[29];

    const int ROWS = Bb * Ss;                 // 8192
    const size_t NX = (size_t)ROWS * Dd;      // 8192*768

    float* ws  = (float*)d_ws;
    float* X   = ws;
    float* Qb  = X + NX;
    float* Kb  = Qb + NX;
    float* Vb  = Kb + NX;
    float* AO  = Vb + NX;
    float* PO  = AO + NX;
    float* FF1 = PO + NX;
    float* EMB = FF1 + (size_t)ROWS * FFd;
    float* XZF = EMB + (size_t)ROWS * EMBd;
    float* XZB = XZF + (size_t)ROWS * GATE;
    float* HF  = XZB + (size_t)ROWS * GATE;
    float* HB  = HF + (size_t)ROWS * HIDd;

    auto gemm = [&](const float* A, const float* Bm, const float* bias, float* Cm,
                    int M, int N, int Kd, int tb, int act) {
        dim3 g((N + 63) / 64, (M + 127) / 128);
        gemm_kernel<<<g, 256, 0, stream>>>(A, Bm, bias, Cm, M, N, Kd, tb, act);
    };

    // embedding + LN
    embed_kernel<<<ROWS, 256, 0, stream>>>(ids, word_emb, pos_emb, X);
    addln_kernel<<<ROWS, 256, 0, stream>>>(X, nullptr, emb_g, emb_b, X, nullptr, 0);

    for (int l = 0; l < Ll; ++l) {
        const size_t wDD = (size_t)l * Dd * Dd;
        gemm(X, Wq + wDD, bq + l * Dd, Qb, ROWS, Dd, Dd, 0, 0);
        gemm(X, Wk + wDD, bk + l * Dd, Kb, ROWS, Dd, Dd, 0, 0);
        gemm(X, Wv + wDD, bv + l * Dd, Vb, ROWS, Dd, Dd, 0, 0);

        attn_kernel<<<Bb * Hh * Cc, 256, 0, stream>>>(Qb, Kb, Vb, amask, AO);

        gemm(AO, Wo + wDD, bo + l * Dd, PO, ROWS, Dd, Dd, 0, 0);
        addln_kernel<<<ROWS, 256, 0, stream>>>(X, PO, g1 + l * Dd, b1 + l * Dd, X, nullptr, 0);

        gemm(X, Wf1 + (size_t)l * Dd * FFd, bf1 + l * FFd, FF1, ROWS, FFd, Dd, 0, 1);  // GELU
        gemm(FF1, Wf2 + (size_t)l * FFd * Dd, bf2 + l * Dd, PO, ROWS, Dd, FFd, 0, 0);
        addln_kernel<<<ROWS, 256, 0, stream>>>(X, PO, g2 + l * Dd, b2 + l * Dd, X, EMB, l * Dd);
    }

    // LSTM input projections: emb @ Wih^T + b  (transB)
    gemm(EMB, Wih_f, b_f, XZF, ROWS, GATE, EMBd, 1, 0);
    gemm(EMB, Wih_b, b_b, XZB, ROWS, GATE, EMBd, 1, 0);

    lstm_kernel<<<1, 256, 0, stream>>>(XZF, Whh_f, HF, 0);
    lstm_kernel<<<1, 256, 0, stream>>>(XZB, Whh_b, HB, 1);

    cls_kernel<<<(ROWS + 255) / 256, 256, 0, stream>>>(HF, HB, clsW, clsb, (float*)d_out);

    (void)in_sizes; (void)n_in; (void)out_size; (void)ws_size;
}
